// CrossAttention_59717225284223
// MI455X (gfx1250) — matrix operand order repaired and checked by reading, not compiled, not run on hardware
//
#include <hip/hip_runtime.h>
#include <hip/hip_bf16.h>
#include <math.h>

#define B_ 4
#define C_ 256
#define N_ 8192
#define H_ 4
#define D_ 64
#define MTOT (B_*N_)   // 32768 positions

typedef __bf16 bf16_t;
typedef __attribute__((ext_vector_type(2)))  bf16_t v2bf;
typedef __attribute__((ext_vector_type(16))) bf16_t v16bf;
typedef __attribute__((ext_vector_type(8)))  float  v8f;

union Frag { v16bf v; unsigned int u[8]; };

// Native f32 -> bf16 (RNE); lets the backend use v_cvt(_pk)_bf16_f32.
__device__ __forceinline__ unsigned short f2bf(float f) {
  union { bf16_t h; unsigned short u; } cv;
  cv.h = (bf16_t)f;
  return cv.u;
}
__device__ __forceinline__ float bf2f(unsigned short h) {
  return __uint_as_float(((unsigned int)h) << 16);
}
__device__ __forceinline__ unsigned int pack2(float a, float b) {
  union { v2bf v; unsigned int u; } cv;
  cv.v[0] = (bf16_t)a;
  cv.v[1] = (bf16_t)b;
  return cv.u;
}

// A fragment (16 rows x K32 window) from LDS tile stored row-major [row][k].
// Contiguous k order used here (same as gatherB): lane l elem e -> row = l%16, K = (l/16)*16 + e.
__device__ __forceinline__ v16bf gatherA(const unsigned short* lds, int rowBase,
                                         int rowStride, int kOff) {
  int lane = threadIdx.x & 31;
  const unsigned int* p =
      (const unsigned int*)(lds + (size_t)(rowBase + (lane & 15)) * rowStride + kOff);
  int o = (lane >> 4) * 8;
  Frag f;
#pragma unroll
  for (int j = 0; j < 4; ++j) { f.u[j] = p[o + j]; f.u[4 + j] = p[o + 4 + j]; }
  return f.v;
}

// B fragment (K32 x 16 cols) from LDS tile stored [col][k].
// Contiguous k order used here (same as gatherA): lane l elem e -> col = l%16, K = (l/16)*16 + e.
__device__ __forceinline__ v16bf gatherB(const unsigned short* lds, int colBase,
                                         int rowStride, int kOff) {
  int lane = threadIdx.x & 31;
  const unsigned int* p =
      (const unsigned int*)(lds + (size_t)(colBase + (lane & 15)) * rowStride + kOff +
                            (lane >> 4) * 16);
  Frag f;
#pragma unroll
  for (int j = 0; j < 8; ++j) f.u[j] = p[j];
  return f.v;
}

__device__ __forceinline__ v8f wmma_bf16(v16bf a, v16bf b, v8f c) {
  return __builtin_amdgcn_wmma_f32_16x16x32_bf16(false, a, false, b, (short)0, c,
                                                 false, false);
}

// ---------------------------------------------------------------------------
// Kernel 1: projection GEMM  out = op(W @ x[b] + bias)
// mode 0: elu+1 -> bf16 [pos][C] ; mode 1: +bias -> bf16 [pos][C]
// mode 2: BN affine -> f32 [b][C][N]
// ---------------------------------------------------------------------------
__global__ __launch_bounds__(256) void proj_gemm(
    const float* __restrict__ x, const float* __restrict__ W,
    const float* __restrict__ bias, unsigned short* __restrict__ outb,
    float* __restrict__ outf, const float* __restrict__ bng,
    const float* __restrict__ bnb, const float* __restrict__ bnm,
    const float* __restrict__ bnv, int mode) {
  __shared__ unsigned short lA[128 * 32];  // weights  [o][k]
  __shared__ unsigned short lB[64 * 32];   // x tile   [n][k]
  const int t = threadIdx.x;
  const int n0 = blockIdx.x * 64;
  const int o0 = blockIdx.y * 128;
  const int b = blockIdx.z;
  const int wave = t >> 5, lane = t & 31;
  const int obase = (wave & 1) * 64;
  const int nbase = (wave >> 1) * 16;
  const float* xb = x + (size_t)b * C_ * N_;
  v8f acc[4] = {};
  for (int ks = 0; ks < 8; ++ks) {
    int c0 = ks * 32;
    if (ks < 7) {  // prefetch next K-slice (global_prefetch_b8)
      __builtin_prefetch(W + (size_t)(o0 + (t >> 1)) * C_ + c0 + 32, 0, 3);
      __builtin_prefetch(xb + (size_t)(c0 + 32 + (t >> 3)) * N_ + n0 + (t & 7) * 8, 0, 3);
    }
#pragma unroll
    for (int j = 0; j < 4; ++j) {  // A: 128x32 weights
      int e = t * 16 + j * 4;
      int row = e >> 5, col = e & 31;
      float4 f4 = *(const float4*)(W + (size_t)(o0 + row) * C_ + c0 + col);
      unsigned int* dst = (unsigned int*)(lA + row * 32 + col);
      dst[0] = pack2(f4.x, f4.y);
      dst[1] = pack2(f4.z, f4.w);
    }
    {  // B: x rows c0..+31, cols n0..+63, transposed into [n][k]
      int c = t >> 3, ng = (t & 7) * 8;
      const float* src = xb + (size_t)(c0 + c) * N_ + n0 + ng;
      float4 f0 = *(const float4*)src;
      float4 f1 = *(const float4*)(src + 4);
      float v[8] = {f0.x, f0.y, f0.z, f0.w, f1.x, f1.y, f1.z, f1.w};
#pragma unroll
      for (int j = 0; j < 8; ++j) lB[(ng + j) * 32 + c] = f2bf(v[j]);
    }
    __syncthreads();
    v16bf bf = gatherB(lB, nbase, 32, 0);
#pragma unroll
    for (int mt = 0; mt < 4; ++mt) {
      v16bf af = gatherA(lA, obase + mt * 16, 32, 0);
      acc[mt] = wmma_bf16(af, bf, acc[mt]);
    }
    __syncthreads();
  }
  int n = n0 + nbase + (lane & 15);
#pragma unroll
  for (int mt = 0; mt < 4; ++mt) {
    int orow0 = o0 + obase + mt * 16 + (lane >> 4) * 8;
    if (mode == 2) {
#pragma unroll
      for (int i = 0; i < 8; ++i) {
        int o = orow0 + i;
        float s = bng[o] * rsqrtf(bnv[o] + 1e-5f);
        outf[(size_t)b * C_ * N_ + (size_t)o * N_ + n] =
            (acc[mt][i] + bias[o] - bnm[o]) * s + bnb[o];
      }
    } else {
      float v[8];
#pragma unroll
      for (int i = 0; i < 8; ++i) {
        float val = acc[mt][i] + bias[orow0 + i];
        if (mode == 0) val = (val > 0.f) ? (val + 1.f) : __expf(val);  // elu+1
        v[i] = val;
      }
      unsigned short* dst = outb + ((size_t)b * N_ + n) * C_ + orow0;
      *(uint2*)(dst)     = make_uint2(pack2(v[0], v[1]), pack2(v[2], v[3]));
      *(uint2*)(dst + 4) = make_uint2(pack2(v[4], v[5]), pack2(v[6], v[7]));
    }
  }
}

// ---------------------------------------------------------------------------
// Kernel 2: per (b,h) KV = K^T @ V over an N-slice (deterministic partials)
// ---------------------------------------------------------------------------
__global__ __launch_bounds__(256) void kv_gemm(
    const unsigned short* __restrict__ Kbf, const unsigned short* __restrict__ Vbf,
    float* __restrict__ KVpart) {
  __shared__ unsigned short lKT[64 * 32];  // [dd][nl]
  __shared__ unsigned short lVT[64 * 32];  // [vv][nl]
  const int t = threadIdx.x;
  const int h = blockIdx.x, b = blockIdx.y, slice = blockIdx.z;
  const int wave = t >> 5, lane = t & 31;
  const int ddt = wave & 3, vvb = (wave >> 2) * 2;
  const int nl = t & 31, g8 = (t >> 5) * 8;
  v8f acc[2] = {};
  for (int n0 = slice * 1024; n0 < slice * 1024 + 1024; n0 += 32) {
    size_t row = ((size_t)b * N_ + n0 + nl) * C_ + h * D_ + g8;
    uint4 kq = *(const uint4*)(Kbf + row);
    uint4 vq = *(const uint4*)(Vbf + row);
    if (n0 + 32 < slice * 1024 + 1024) {
      __builtin_prefetch(Kbf + row + 32 * C_, 0, 3);
      __builtin_prefetch(Vbf + row + 32 * C_, 0, 3);
    }
    unsigned short ksv[8], vsv[8];
    *(uint4*)ksv = kq;
    *(uint4*)vsv = vq;
#pragma unroll
    for (int j = 0; j < 8; ++j) {
      lKT[(g8 + j) * 32 + nl] = ksv[j];
      lVT[(g8 + j) * 32 + nl] = vsv[j];
    }
    __syncthreads();
    v16bf af = gatherA(lKT, ddt * 16, 32, 0);
#pragma unroll
    for (int j = 0; j < 2; ++j) {
      v16bf bf = gatherB(lVT, (vvb + j) * 16, 32, 0);
      acc[j] = wmma_bf16(af, bf, acc[j]);
    }
    __syncthreads();
  }
#pragma unroll
  for (int j = 0; j < 2; ++j)
#pragma unroll
    for (int i = 0; i < 8; ++i) {
      int dd = ddt * 16 + (lane >> 4) * 8 + i;
      int vv = (vvb + j) * 16 + (lane & 15);
      KVpart[(size_t)slice * (B_ * H_ * 4096) + ((size_t)b * H_ + h) * 4096 +
             dd * 64 + vv] = acc[j][i];
    }
}

__global__ __launch_bounds__(256) void kv_reduce(const float* __restrict__ KVpart,
                                                 float* __restrict__ KV) {
  int i = blockIdx.x * 256 + threadIdx.x;  // 65536 total
  float s = 0.f;
#pragma unroll
  for (int p = 0; p < 8; ++p) s += KVpart[(size_t)p * 65536 + i];
  KV[i] = s;
}

__global__ __launch_bounds__(256) void ksum_kernel(
    const unsigned short* __restrict__ Kbf, float* __restrict__ Ksum) {
  __shared__ float red[256];
  int bh = blockIdx.x, b = bh >> 2, h = bh & 3;
  int dd = threadIdx.x & 63, slice = threadIdx.x >> 6;
  float s = 0.f;
  for (int n = slice * (N_ / 4); n < (slice + 1) * (N_ / 4); ++n)
    s += bf2f(Kbf[((size_t)b * N_ + n) * C_ + h * D_ + dd]);
  red[threadIdx.x] = s;
  __syncthreads();
  if (slice == 0)
    Ksum[(size_t)bh * D_ + dd] = red[dd] + red[64 + dd] + red[128 + dd] + red[192 + dd];
}

// ---------------------------------------------------------------------------
// Kernel 3: scores -> Z, msg = (Q @ KV) * Z, LayerNorm(g1,b1). 32 pos/block.
// ---------------------------------------------------------------------------
__global__ __launch_bounds__(256) void attn_msg_kernel(
    const unsigned short* __restrict__ Qbf, const float* __restrict__ KV,
    const float* __restrict__ Ksum, const float* __restrict__ thrp,
    const float* __restrict__ g1, const float* __restrict__ b1,
    float* __restrict__ msgLN) {
  __shared__ unsigned int smem[12288];  // 48KB union
  unsigned short* lQ = (unsigned short*)smem;           // 16KB [pos][256]
  unsigned short* lKV = (unsigned short*)smem + 8192;   // 32KB [h][vv][dd]
  float* lMsg = (float*)smem;                           // 32KB [pos][256] (reused)
  __shared__ float lKsum[256];
  __shared__ float lZ[32 * 4];
  const int t = threadIdx.x;
  const int n0 = blockIdx.x * 32;
  const int b = blockIdx.y;
  const int wave = t >> 5, lane = t & 31;
  // stage Q (bf16 copy)
  const unsigned int* qsrc = (const unsigned int*)(Qbf + ((size_t)b * N_ + n0) * C_);
#pragma unroll
  for (int j = 0; j < 16; ++j) ((unsigned int*)lQ)[t + j * 256] = qsrc[t + j * 256];
  // stage KV transposed -> [h][vv][dd] bf16
  for (int j = 0; j < 64; ++j) {
    int m = t + j * 256;
    int h = m >> 12, r = m & 4095, vv = r >> 6, dd = r & 63;
    lKV[m] = f2bf(KV[(((size_t)b * H_ + h) * 64 + dd) * 64 + vv]);
  }
  lKsum[t] = Ksum[(size_t)b * 256 + t];
  __syncthreads();
  if (t < 128) {  // scores + Z
    int pos = t & 31, h = t >> 5;
    float s = 0.f;
    for (int dd = 0; dd < 64; ++dd)
      s += bf2f(lQ[pos * 256 + h * 64 + dd]) * lKsum[h * 64 + dd];
    float thr = thrp[0];
    float sp = (s > thr) ? s : 0.f;
    lZ[pos * 4 + h] = 1.f / (sp + 1e-6f);
  }
  __syncthreads();
  // msg = Q @ KV per head (K = 64 -> two k-steps)
  const int rt = wave & 1, h = wave >> 1;
  v8f acc[4] = {};
#pragma unroll
  for (int s = 0; s < 2; ++s) {
    v16bf af = gatherA(lQ, rt * 16, 256, h * 64 + s * 32);
#pragma unroll
    for (int vt = 0; vt < 4; ++vt) {
      v16bf bf = gatherB(lKV, h * 64 + vt * 16, 64, s * 32);
      acc[vt] = wmma_bf16(af, bf, acc[vt]);
    }
  }
  __syncthreads();  // all gathers done; lQ/lKV now dead
#pragma unroll
  for (int vt = 0; vt < 4; ++vt)
#pragma unroll
    for (int i = 0; i < 8; ++i) {
      int pos = rt * 16 + (lane >> 4) * 8 + i;
      int c = h * 64 + vt * 16 + (lane & 15);
      lMsg[pos * 256 + c] = acc[vt][i] * lZ[pos * 4 + h];
    }
  __syncthreads();
  // LayerNorm per row (wave per row)
  for (int it = 0; it < 4; ++it) {
    int row = it * 8 + wave;
    float xv[8], s = 0.f, q = 0.f;
#pragma unroll
    for (int j = 0; j < 8; ++j) {
      xv[j] = lMsg[row * 256 + lane + j * 32];
      s += xv[j];
      q += xv[j] * xv[j];
    }
#pragma unroll
    for (int off = 16; off > 0; off >>= 1) {
      s += __shfl_xor(s, off, 32);
      q += __shfl_xor(q, off, 32);
    }
    float mean = s * (1.f / 256.f);
    float var = q * (1.f / 256.f) - mean * mean;
    float inv = rsqrtf(var + 1e-5f);
#pragma unroll
    for (int j = 0; j < 8; ++j) {
      int c = lane + j * 32;
      msgLN[((size_t)b * N_ + n0 + row) * C_ + c] =
          (xv[j] - mean) * inv * g1[c] + b1[c];
    }
  }
}

// ---------------------------------------------------------------------------
// Kernel 4: h1 = relu([x1^T | msgLN] @ W1^T), bf16 out [pos][512]
// ---------------------------------------------------------------------------
__global__ __launch_bounds__(256) void mlp1_kernel(
    const float* __restrict__ x1, const float* __restrict__ msgLN,
    const float* __restrict__ W1, unsigned short* __restrict__ h1) {
  __shared__ unsigned short lA[64 * 32];
  __shared__ unsigned short lB[64 * 32];
  const int t = threadIdx.x;
  const int gpos0 = blockIdx.x * 64;
  const int j0 = blockIdx.y * 64;
  const int b = gpos0 >> 13, n0 = gpos0 & (N_ - 1);
  const int wave = t >> 5, lane = t & 31;
  const int rt = wave & 3, cp = (wave >> 2) * 2;
  v8f acc[2] = {};
  for (int ks = 0; ks < 16; ++ks) {
    int k0 = ks * 32;
    if (ks < 15)  // prefetch next W1 K-slice
      __builtin_prefetch(W1 + (size_t)(j0 + (t >> 2)) * 512 + k0 + 32, 0, 3);
    if (k0 < 256) {  // A from x1 (strided channels, contiguous n)
      int kl = t >> 3, pg = (t & 7) * 8;
      const float* src = x1 + (size_t)b * C_ * N_ + (size_t)(k0 + kl) * N_ + n0 + pg;
      float4 f0 = *(const float4*)src;
      float4 f1 = *(const float4*)(src + 4);
      float v[8] = {f0.x, f0.y, f0.z, f0.w, f1.x, f1.y, f1.z, f1.w};
#pragma unroll
      for (int j = 0; j < 8; ++j) lA[(pg + j) * 32 + kl] = f2bf(v[j]);
    } else {  // A from msgLN rows
      int pl = t >> 2, kg = (t & 3) * 8;
      const float* src = msgLN + (size_t)(gpos0 + pl) * C_ + (k0 - 256) + kg;
      float4 f0 = *(const float4*)src;
      float4 f1 = *(const float4*)(src + 4);
      unsigned int* dst = (unsigned int*)(lA + pl * 32 + kg);
      dst[0] = pack2(f0.x, f0.y); dst[1] = pack2(f0.z, f0.w);
      dst[2] = pack2(f1.x, f1.y); dst[3] = pack2(f1.z, f1.w);
    }
    {  // B from W1 rows (out j, in k)
      int jl = t >> 2, kg = (t & 3) * 8;
      const float* src = W1 + (size_t)(j0 + jl) * 512 + k0 + kg;
      float4 f0 = *(const float4*)src;
      float4 f1 = *(const float4*)(src + 4);
      unsigned int* dst = (unsigned int*)(lB + jl * 32 + kg);
      dst[0] = pack2(f0.x, f0.y); dst[1] = pack2(f0.z, f0.w);
      dst[2] = pack2(f1.x, f1.y); dst[3] = pack2(f1.z, f1.w);
    }
    __syncthreads();
    v16bf af = gatherA(lA, rt * 16, 32, 0);
#pragma unroll
    for (int j = 0; j < 2; ++j) {
      v16bf bf = gatherB(lB, (cp + j) * 16, 32, 0);
      acc[j] = wmma_bf16(af, bf, acc[j]);
    }
    __syncthreads();
  }
#pragma unroll
  for (int j = 0; j < 2; ++j)
#pragma unroll
    for (int i = 0; i < 8; ++i) {
      int pos = gpos0 + rt * 16 + (lane >> 4) * 8 + i;
      int col = j0 + (cp + j) * 16 + (lane & 15);
      float v = acc[j][i];
      h1[(size_t)pos * 512 + col] = f2bf(v > 0.f ? v : 0.f);
    }
}

// ---------------------------------------------------------------------------
// Kernel 5: m2 = h1 @ W2^T, LayerNorm(g2,b2), out = m2 + x1 + aug (transposed)
// ---------------------------------------------------------------------------
__global__ __launch_bounds__(256) void mlp2_final_kernel(
    const unsigned short* __restrict__ h1, const float* __restrict__ W2,
    const float* __restrict__ g2, const float* __restrict__ b2,
    const float* __restrict__ x1, const float* __restrict__ aug,
    float* __restrict__ out) {
  __shared__ unsigned int smem[16384];  // 64KB union
  unsigned short* lA = (unsigned short*)smem;          // 4KB [64 pos][32]
  unsigned short* lB = (unsigned short*)smem + 2048;   // 16KB [256 j][32]
  float* lMsg = (float*)smem;                          // 64KB [64][256] (reused)
  const int t = threadIdx.x;
  const int gpos0 = blockIdx.x * 64;
  const int b = gpos0 >> 13, n0 = gpos0 & (N_ - 1);
  const int wave = t >> 5, lane = t & 31;
  const int rt = wave >> 1, ch = wave & 1;
  v8f acc[8] = {};
  for (int ks = 0; ks < 16; ++ks) {
    int k0 = ks * 32;
    if (ks < 15)  // prefetch next W2 K-slice
      __builtin_prefetch(W2 + (size_t)t * 512 + k0 + 32, 0, 3);
    {  // A from h1 (already bf16)
      int pl = t >> 2, kg = (t & 3) * 8;
      uint4 q = *(const uint4*)(h1 + (size_t)(gpos0 + pl) * 512 + k0 + kg);
      *(uint4*)(lA + pl * 32 + kg) = q;
    }
    {  // B: all 256 W2 rows, 32-wide k window
      const float* src = W2 + (size_t)t * 512 + k0;
      unsigned int* dst = (unsigned int*)(lB + t * 32);
#pragma unroll
      for (int g = 0; g < 4; ++g) {
        float4 f0 = *(const float4*)(src + g * 8);
        float4 f1 = *(const float4*)(src + g * 8 + 4);
        dst[g * 4 + 0] = pack2(f0.x, f0.y); dst[g * 4 + 1] = pack2(f0.z, f0.w);
        dst[g * 4 + 2] = pack2(f1.x, f1.y); dst[g * 4 + 3] = pack2(f1.z, f1.w);
      }
    }
    __syncthreads();
    v16bf af = gatherA(lA, rt * 16, 32, 0);
#pragma unroll
    for (int j = 0; j < 8; ++j) {
      v16bf bf = gatherB(lB, ch * 128 + j * 16, 32, 0);
      acc[j] = wmma_bf16(af, bf, acc[j]);
    }
    __syncthreads();
  }
#pragma unroll
  for (int j = 0; j < 8; ++j)
#pragma unroll
    for (int i = 0; i < 8; ++i) {
      int pl = rt * 16 + (lane >> 4) * 8 + i;
      int c = ch * 128 + j * 16 + (lane & 15);
      lMsg[pl * 256 + c] = acc[j][i];
    }
  __syncthreads();
  for (int it = 0; it < 8; ++it) {  // LayerNorm, wave per row
    int row = it * 8 + wave;
    float xv[8], s = 0.f, q = 0.f;
#pragma unroll
    for (int j = 0; j < 8; ++j) {
      xv[j] = lMsg[row * 256 + lane + j * 32];
      s += xv[j];
      q += xv[j] * xv[j];
    }
#pragma unroll
    for (int off = 16; off > 0; off >>= 1) {
      s += __shfl_xor(s, off, 32);
      q += __shfl_xor(q, off, 32);
    }
    float mean = s * (1.f / 256.f);
    float var = q * (1.f / 256.f) - mean * mean;
    float inv = rsqrtf(var + 1e-5f);
#pragma unroll
    for (int j = 0; j < 8; ++j) {
      int c = lane + j * 32;
      lMsg[row * 256 + c] = (xv[j] - mean) * inv * g2[c] + b2[c];
    }
  }
  __syncthreads();
  // transposed, coalesced final write: out = msg + x1 + aug
  int nl = t & 63, cb = (t >> 6) * 64;
  for (int cc = 0; cc < 64; ++cc) {
    int c = cb + cc;
    size_t addr = (size_t)b * C_ * N_ + (size_t)c * N_ + n0 + nl;
    out[addr] = lMsg[nl * 256 + c] + x1[addr] + aug[addr];
  }
}

extern "C" void kernel_launch(void* const* d_in, const int* in_sizes, int n_in,
                              void* d_out, int out_size, void* d_ws, size_t ws_size,
                              hipStream_t stream) {
  (void)in_sizes; (void)n_in; (void)out_size; (void)ws_size;
  const float* x1 = (const float*)d_in[0];
  const float* x2 = (const float*)d_in[1];
  const float* Wq = (const float*)d_in[2];
  const float* bq = (const float*)d_in[3];
  const float* Wk = (const float*)d_in[4];
  const float* bk = (const float*)d_in[5];
  const float* Wv = (const float*)d_in[6];
  const float* bv = (const float*)d_in[7];
  const float* thr = (const float*)d_in[8];
  const float* W1 = (const float*)d_in[9];
  const float* W2 = (const float*)d_in[10];
  const float* g1 = (const float*)d_in[11];
  const float* b1 = (const float*)d_in[12];
  const float* g2 = (const float*)d_in[13];
  const float* b2 = (const float*)d_in[14];
  const float* Wa = (const float*)d_in[15];
  const float* ba = (const float*)d_in[16];
  const float* bn_g = (const float*)d_in[17];
  const float* bn_b = (const float*)d_in[18];
  const float* bn_m = (const float*)d_in[19];
  const float* bn_v = (const float*)d_in[20];
  float* out = (float*)d_out;

  char* ws = (char*)d_ws;
  size_t off = 0;
  auto alloc = [&](size_t bytes) {
    char* p = ws + off;
    off += (bytes + 255) & ~(size_t)255;
    return p;
  };
  unsigned short* Qbf = (unsigned short*)alloc((size_t)MTOT * C_ * 2);  // 16MB
  unsigned short* Kbf = (unsigned short*)alloc((size_t)MTOT * C_ * 2);  // 16MB
  unsigned short* Vbf = (unsigned short*)alloc((size_t)MTOT * C_ * 2);  // 16MB
  float* aug = (float*)alloc((size_t)MTOT * C_ * 4);                    // 32MB
  float* msgLN = (float*)alloc((size_t)MTOT * C_ * 4);                  // 32MB
  float* KVpart = (float*)alloc((size_t)8 * B_ * H_ * 4096 * 4);        // 2MB
  float* KV = (float*)alloc((size_t)B_ * H_ * 4096 * 4);
  float* Ksum = (float*)alloc((size_t)B_ * H_ * 64 * 4);
  // h1 reuses Q+K region (dead after attn_msg_kernel): 32768*512*2 = 32MB
  unsigned short* h1 = Qbf;

  dim3 blk(256);
  dim3 gproj(N_ / 64, C_ / 128, B_);
  proj_gemm<<<gproj, blk, 0, stream>>>(x1, Wq, bq, Qbf, nullptr, nullptr, nullptr,
                                       nullptr, nullptr, 0);
  proj_gemm<<<gproj, blk, 0, stream>>>(x2, Wk, bk, Kbf, nullptr, nullptr, nullptr,
                                       nullptr, nullptr, 0);
  proj_gemm<<<gproj, blk, 0, stream>>>(x2, Wv, bv, Vbf, nullptr, nullptr, nullptr,
                                       nullptr, nullptr, 1);
  proj_gemm<<<gproj, blk, 0, stream>>>(x1, Wa, ba, nullptr, aug, bn_g, bn_b, bn_m,
                                       bn_v, 2);
  kv_gemm<<<dim3(H_, B_, 8), blk, 0, stream>>>(Kbf, Vbf, KVpart);
  kv_reduce<<<dim3(256), blk, 0, stream>>>(KVpart, KV);
  ksum_kernel<<<dim3(B_ * H_), blk, 0, stream>>>(Kbf, Ksum);
  attn_msg_kernel<<<dim3(N_ / 32, B_), blk, 0, stream>>>(Qbf, KV, Ksum, thr, g1, b1,
                                                         msgLN);
  mlp1_kernel<<<dim3(MTOT / 64, 512 / 64), blk, 0, stream>>>(x1, msgLN, W1, h1);
  mlp2_final_kernel<<<dim3(MTOT / 64), blk, 0, stream>>>(h1, W2, g2, b2, x1, aug, out);
}
